// smooth_network_57114475102675
// MI455X (gfx1250) — 16-bit and 8-bit matrix operand order repaired and checked by reading, compiled, not run on hardware
//
#include <hip/hip_runtime.h>

typedef __attribute__((ext_vector_type(16))) _Float16 v16h;
typedef __attribute__((ext_vector_type(8)))  _Float16 v8h;
typedef __attribute__((ext_vector_type(4)))  _Float16 v4h;
typedef __attribute__((ext_vector_type(8)))  float    v8f;
typedef __attribute__((ext_vector_type(4)))  float    v4f;
typedef __attribute__((ext_vector_type(8)))  int      v8i;
typedef __attribute__((ext_vector_type(4)))  int      v4i;
typedef __attribute__((ext_vector_type(2)))  int      v2i;

#define NWIN 2048
#define SEQ  64
#define DIM  256
#define NGRP 8
#define QMAXF 127.0f

#define XS_STRIDE 264
#define QS_STRIDE 272


__global__ void k_init(unsigned int* maxes) {
    if (threadIdx.x < 2) maxes[threadIdx.x] = 0u;
}

__global__ void k_prep(const float* __restrict__ A, const float* __restrict__ W,
                       _Float16* __restrict__ AT, _Float16* __restrict__ WT) {
    int i = blockIdx.x * 256 + threadIdx.x;
    const int nA = NGRP * DIM * DIM;
    if (i < nA) {
        int g = i / (DIM * DIM);
        int r = i % (DIM * DIM);
        int n = r / DIM, k = r % DIM;
        AT[i] = (_Float16)A[g * DIM * DIM + k * DIM + n];
    } else {
        int j = i - nA;
        if (j < DIM * DIM) {
            int n = j / DIM, k = j % DIM;
            WT[j] = (_Float16)W[k * DIM + n];
        }
    }
}

template <bool F16OUT>
__device__ __forceinline__ void gemm64x256_f16(
    const float* __restrict__ Asrc,
    const _Float16* __restrict__ BT,
    float* __restrict__ outF32,
    _Float16* __restrict__ outF16,
    unsigned int* __restrict__ maxSlot)
{
    __shared__ _Float16 xs[SEQ * XS_STRIDE];
    __shared__ float smax[8];

#pragma unroll
    for (int it = 0; it < 16; ++it) {
        int row = it * 4 + (threadIdx.x >> 6);
        int c4  = (threadIdx.x & 63) * 4;
        v4f v = *(const v4f*)(Asrc + row * DIM + c4);
        v4h h;
        h[0] = (_Float16)v[0]; h[1] = (_Float16)v[1];
        h[2] = (_Float16)v[2]; h[3] = (_Float16)v[3];
        *(v4h*)(&xs[row * XS_STRIDE + c4]) = h;
    }
    __syncthreads();

    const int lane = threadIdx.x & 31;
    const int wave = threadIdx.x >> 5;
    const int half = lane >> 4;
    const int l16  = lane & 15;
    const int mt   = wave & 3;
    const int ntB  = (wave >> 2) * 8;
    const int arow = mt * 16 + l16;

    v8f acc[8] = {};

    for (int ks = 0; ks < 8; ++ks) {
        const _Float16* ap = &xs[arow * XS_STRIDE + ks * 32 + half * 16];
        v8h lo = *(const v8h*)ap;
        v8h hi = *(const v8h*)(ap + 8);
        v16h a;
#pragma unroll
        for (int j = 0; j < 8; ++j) { a[j] = lo[j]; a[8 + j] = hi[j]; }

        if (ks < 7)
            __builtin_prefetch((const void*)(BT + (ntB * 16 + l16) * DIM + (ks + 1) * 32), 0, 1);
#pragma unroll
        for (int nt = 0; nt < 8; ++nt) {
            int n = (ntB + nt) * 16 + l16;
            v16h b = *(const v16h*)(BT + n * DIM + ks * 32 + half * 16);
            acc[nt] = __builtin_amdgcn_wmma_f32_16x16x32_f16(
                false, a, false, b, (short)0, acc[nt], false, false);
        }
    }

    float m = 0.0f;
#pragma unroll
    for (int nt = 0; nt < 8; ++nt) {
        int col = (ntB + nt) * 16 + l16;
#pragma unroll
        for (int r = 0; r < 8; ++r) {
            int orow = mt * 16 + r + 8 * half;
            float v = acc[nt][r];
            if (F16OUT) {
                _Float16 h = (_Float16)v;
                outF16[orow * DIM + col] = h;
                m = fmaxf(m, fabsf((float)h));
            } else {
                outF32[orow * DIM + col] = v;
                m = fmaxf(m, fabsf(v));
            }
        }
    }
#pragma unroll
    for (int off = 16; off > 0; off >>= 1)
        m = fmaxf(m, __shfl_xor(m, off, 32));
    if (lane == 0) smax[wave] = m;
    __syncthreads();
    if (threadIdx.x == 0) {
        float mm = smax[0];
#pragma unroll
        for (int i = 1; i < 8; ++i) mm = fmaxf(mm, smax[i]);
        atomicMax(maxSlot, __float_as_uint(mm));
    }
}

__global__ void k_bw(const float* __restrict__ Bm, const _Float16* __restrict__ WT,
                     float* __restrict__ BW, unsigned int* __restrict__ maxSlot) {
    int g  = blockIdx.x >> 2;
    int mc = blockIdx.x & 3;
    gemm64x256_f16<false>(Bm + g * DIM * DIM + mc * 64 * DIM, WT,
                          BW + g * DIM * DIM + mc * 64 * DIM, nullptr, maxSlot);
}

__global__ void k_xa(const float* __restrict__ X, const _Float16* __restrict__ AT,
                     const int* __restrict__ labels,
                     _Float16* __restrict__ XA, unsigned int* __restrict__ maxSlot) {
    int w = blockIdx.x;
    int g = labels[w];
    gemm64x256_f16<true>(X + w * SEQ * DIM, AT + g * DIM * DIM,
                         nullptr, XA + w * SEQ * DIM, maxSlot);
}

__global__ void k_qbw(const float* __restrict__ BW,
                      const unsigned int* __restrict__ maxes,
                      signed char* __restrict__ BWq) {
    float scale = fmaxf(__uint_as_float(maxes[1]) / QMAXF, 1e-8f);
    float inv = 1.0f / scale;
    int i = blockIdx.x * 256 + threadIdx.x;
    int g = i / (DIM * DIM);
    int r = i % (DIM * DIM);
    int n = r / DIM, k = r % DIM;
    float q = fminf(fmaxf(rintf(BW[g * DIM * DIM + k * DIM + n] * inv), -128.0f), 127.0f);
    BWq[i] = (signed char)q;
}

__global__ void k_out(const _Float16* __restrict__ XA,
                      const signed char* __restrict__ BWq,
                      const int* __restrict__ labels,
                      const unsigned int* __restrict__ maxes,
                      float* __restrict__ out) {
    __shared__ signed char qs[SEQ * QS_STRIDE];

    int w = blockIdx.x;
    int g = labels[w];
    float sA = fmaxf(__uint_as_float(maxes[0]) / QMAXF, 1e-8f);
    float sB = fmaxf(__uint_as_float(maxes[1]) / QMAXF, 1e-8f);
    float invA = 1.0f / sA;
    float s2 = sA * sB;

    const _Float16*    Xw = XA  + (size_t)w * SEQ * DIM;
    const signed char* Bg = BWq + (size_t)g * DIM * DIM;

#pragma unroll
    for (int it = 0; it < 8; ++it) {
        int row = it * 8 + (threadIdx.x >> 5);
        int c8  = (threadIdx.x & 31) * 8;
        v8h hv = *(const v8h*)(Xw + row * DIM + c8);
        int p0 = 0, p1 = 0;
#pragma unroll
        for (int j = 0; j < 4; ++j) {
            int q0 = (int)fminf(fmaxf(rintf((float)hv[j]     * invA), -128.0f), 127.0f);
            int q1 = (int)fminf(fmaxf(rintf((float)hv[4 + j] * invA), -128.0f), 127.0f);
            p0 |= (q0 & 255) << (8 * j);
            p1 |= (q1 & 255) << (8 * j);
        }
        v2i p; p[0] = p0; p[1] = p1;
        *(v2i*)(&qs[row * QS_STRIDE + c8]) = p;
    }
    __syncthreads();

    const int lane = threadIdx.x & 31;
    const int wave = threadIdx.x >> 5;
    const int half = lane >> 4;
    const int l16  = lane & 15;
    const int mt   = wave & 3;
    const int ntB  = (wave >> 2) * 8;
    const int arow = mt * 16 + l16;

    v8i acc[8] = {};

#pragma unroll
    for (int ks = 0; ks < 4; ++ks) {
        const signed char* ap = &qs[arow * QS_STRIDE + ks * 64 + half * 16];
        v8i a;
#pragma unroll
        for (int c = 0; c < 4; ++c) {
            v2i d = *(const v2i*)(ap + c * 8 + (c >> 1) * 16);
            a[2 * c]     = d[0];
            a[2 * c + 1] = d[1];
        }
#pragma unroll
        for (int nt = 0; nt < 8; ++nt) {
            int n = (ntB + nt) * 16 + l16;
            const signed char* bp = Bg + n * DIM + ks * 64 + half * 16;
            v4i blo = *(const v4i*)bp;
            v4i bhi = *(const v4i*)(bp + 32);
            v8i b;
            b[0] = blo[0]; b[1] = blo[1]; b[2] = blo[2]; b[3] = blo[3];
            b[4] = bhi[0]; b[5] = bhi[1]; b[6] = bhi[2]; b[7] = bhi[3];
            acc[nt] = __builtin_amdgcn_wmma_i32_16x16x64_iu8(
                true, a, true, b, acc[nt], false, false);
        }
    }

    float* ow = out + (size_t)w * SEQ * DIM;
#pragma unroll
    for (int nt = 0; nt < 8; ++nt) {
        int col = (ntB + nt) * 16 + l16;
#pragma unroll
        for (int r = 0; r < 8; ++r) {
            int orow = mt * 16 + r + 8 * half;
            ow[orow * DIM + col] = (float)acc[nt][r] * s2;
        }
    }
}

extern "C" void kernel_launch(void* const* d_in, const int* in_sizes, int n_in,
                              void* d_out, int out_size, void* d_ws, size_t ws_size,
                              hipStream_t stream) {
    const float* X      = (const float*)d_in[0];
    const float* W      = (const float*)d_in[1];
    const float* A      = (const float*)d_in[2];
    const float* Bm     = (const float*)d_in[3];
    const int*   labels = (const int*)d_in[4];
    float* out = (float*)d_out;

    char* ws = (char*)d_ws;
    _Float16*     XA    = (_Float16*)(ws);
    float*        BW    = (float*)(ws + 67108864);
    signed char*  BWq   = (signed char*)(ws + 69206016);
    _Float16*     AT    = (_Float16*)(ws + 69730304);
    _Float16*     WT    = (_Float16*)(ws + 70778880);
    unsigned int* maxes = (unsigned int*)(ws + 70909952);

    k_init<<<1, 32, 0, stream>>>(maxes);
    k_prep<<<(NGRP * DIM * DIM + DIM * DIM + 255) / 256, 256, 0, stream>>>(A, W, AT, WT);
    k_bw <<<NGRP * 4, 256, 0, stream>>>(Bm, WT, BW, maxes + 1);
    k_qbw<<<NGRP * DIM * DIM / 256, 256, 0, stream>>>(BW, maxes, BWq);
    k_xa <<<NWIN, 256, 0, stream>>>(X, AT, labels, XA, maxes + 0);
    k_out<<<NWIN, 256, 0, stream>>>(XA, BWq, labels, maxes, out);
}
